// DecoderV1_19267223290699
// MI455X (gfx1250) — matrix operand order repaired and checked by reading, not compiled, not run on hardware
//
#include <hip/hip_runtime.h>
#include <stdint.h>

typedef __attribute__((ext_vector_type(16))) __bf16 v16bf;
typedef __attribute__((ext_vector_type(8)))  float  v8f;

#define BM 128
#define BN 128
#define BK 32          // bf16 K-depth per WMMA
#define LDT 20         // LDS row pitch in dwords (= 40 bf16 = 80B, 16B aligned, conflict-free)

// ---------------------------------------------------------------------------
// Kernel 1: per-graph mean pool + gather of the two query rows, fused with
// bf16 hi/lo split-precision decomposition of final_embs [B x 3D].
// One block per graph: streams 64x512 f32 (coalesced across threads per row).
// ---------------------------------------------------------------------------
__global__ __launch_bounds__(256)
void build_final_kernel(const float* __restrict__ emb,
                        const int* __restrict__ targets,
                        __bf16* __restrict__ Ahi, __bf16* __restrict__ Alo,
                        int NPER, int D) {
  int b = blockIdx.x;
  const float* g = emb + (size_t)b * NPER * D;
  int t0 = targets[2 * b];
  int t1 = targets[2 * b + 1];
  const float* q0 = g + (size_t)t0 * D;
  const float* q1 = g + (size_t)t1 * D;
  __bf16* ah = Ahi + (size_t)b * 3 * D;
  __bf16* al = Alo + (size_t)b * 3 * D;
  float inv = 1.0f / (float)NPER;

  for (int c = threadIdx.x; c < D; c += blockDim.x) {
    float s = 0.0f;
#pragma unroll 8
    for (int r = 0; r < NPER; ++r) s += g[(size_t)r * D + c];
    float m  = s * inv;
    float x0 = q0[c];
    float x1 = q1[c];
    __bf16 h0 = (__bf16)x0; ah[c]         = h0; al[c]         = (__bf16)(x0 - (float)h0);
    __bf16 h1 = (__bf16)x1; ah[D + c]     = h1; al[D + c]     = (__bf16)(x1 - (float)h1);
    __bf16 hm = (__bf16)m;  ah[2 * D + c] = hm; al[2 * D + c] = (__bf16)(m  - (float)hm);
  }
}

// ---------------------------------------------------------------------------
// Kernel 2: elementwise bf16 hi/lo decomposition of W [T x 3D].
// ---------------------------------------------------------------------------
__global__ __launch_bounds__(256)
void decompose_mat_kernel(const float* __restrict__ src,
                          __bf16* __restrict__ hi, __bf16* __restrict__ lo, int n) {
  int i = blockIdx.x * blockDim.x + threadIdx.x;
  int stride = gridDim.x * blockDim.x;
  for (; i < n; i += stride) {
    float x = src[i];
    __bf16 h = (__bf16)x;
    hi[i] = h;
    lo[i] = (__bf16)(x - (float)h);
  }
}

// ---------------------------------------------------------------------------
// Async global->LDS copy of one 16B chunk per lane (GVS addressing:
// 64-bit uniform SGPR base + 32-bit per-lane byte offset). ASYNCcnt-tracked.
// ---------------------------------------------------------------------------
__device__ __forceinline__ void async_ld16(unsigned lds_off, const void* base,
                                           unsigned byte_off) {
  asm volatile("global_load_async_to_lds_b128 %0, %1, %2"
               :
               : "v"(lds_off), "v"(byte_off), "s"(base)
               : "memory");
}

__device__ __forceinline__ void wait_async0() {
  asm volatile("s_wait_asynccnt 0x0" ::: "memory");
}

// ---------------------------------------------------------------------------
// LDS fragment loaders (wave32 WMMA 16x16x32 bf16 layouts, ISA 7.12.2).
// A-matrix: lane<16 -> M=lane, K in {0..7} U {16..23}; lane>=16 -> K+8.
// B-matrix: lane<16 -> N=lane, K in {0..7} U {16..23}; lane>=16 -> K+8.
// LDS tile stored row-major [row][40 bf16] (LDT=20 dwords pitch).
// ---------------------------------------------------------------------------
__device__ __forceinline__ v16bf frag_a(const unsigned int* lds, int r, int half) {
  v16bf f;
  uint4* fp = (uint4*)&f;
  const unsigned int* p = lds + r * LDT;
  fp[0] = *(const uint4*)(p + half * 4);        // K = half*8 .. +7
  fp[1] = *(const uint4*)(p + 8 + half * 4);    // K = 16 + half*8 .. +7
  return f;
}

__device__ __forceinline__ v16bf frag_b(const unsigned int* lds, int r, int half) {
  v16bf f;
  uint4* fp = (uint4*)&f;
  const unsigned int* p = lds + r * LDT;
  fp[0] = *(const uint4*)(p + half * 4);        // K = half*8 .. +7
  fp[1] = *(const uint4*)(p + 8 + half * 4);    // K = 16 + half*8 .. +7
  return f;
}

// ---------------------------------------------------------------------------
// Kernel 3: out[M x N] = A[M x K] * W[N x K]^T + bias, split-precision bf16x3
// with f32 accumulation. 128x128 block tile, 8 waves of 2x4, 64x32 per wave.
// Double-buffered LDS fed by async global->LDS DMA (overlaps with WMMA).
// ---------------------------------------------------------------------------
__global__ __launch_bounds__(256)
void gemm_bf16x3_kernel(const __bf16* __restrict__ A_hi, const __bf16* __restrict__ A_lo,
                        const __bf16* __restrict__ B_hi, const __bf16* __restrict__ B_lo,
                        const float* __restrict__ bias, float* __restrict__ out,
                        int M, int N, int K) {
  __shared__ __align__(16) unsigned int s_ah[2][BM * LDT];
  __shared__ __align__(16) unsigned int s_al[2][BM * LDT];
  __shared__ __align__(16) unsigned int s_bh[2][BN * LDT];
  __shared__ __align__(16) unsigned int s_bl[2][BN * LDT];

  int t = threadIdx.x;
  int m_blk = blockIdx.y * BM;
  int n_blk = blockIdx.x * BN;
  int wave = t >> 5, lane = t & 31;
  int wm = wave >> 2;       // 0..1 : 64-row strip
  int wn = wave & 3;        // 0..3 : 32-col strip
  int half = lane >> 4;
  int l16  = lane & 15;

  v8f acc[4][2];
  v8f zero = {0.f, 0.f, 0.f, 0.f, 0.f, 0.f, 0.f, 0.f};
#pragma unroll
  for (int i = 0; i < 4; ++i)
#pragma unroll
    for (int j = 0; j < 2; ++j) acc[i][j] = zero;

  int seg  = t & 3;         // which 8-bf16 (16B) chunk of the 32-wide K tile
  int row0 = t >> 2;        // 0..63
  int kt_count = K / BK;

  // Issue the async staging of K-tile `kt` into LDS buffer `buf` (8 b128 DMAs).
  auto issue_stage = [&](int kt, int buf) {
    int kc = kt * BK + seg * 8;   // bf16 column of this thread's chunk
#pragma unroll
    for (int rr = 0; rr < 2; ++rr) {
      int row = row0 + rr * 64;
      unsigned ga = ((unsigned)(m_blk + row) * (unsigned)K + (unsigned)kc) * 2u;
      unsigned gb = ((unsigned)(n_blk + row) * (unsigned)K + (unsigned)kc) * 2u;
      unsigned ldso = (unsigned)(row * LDT + seg * 4) * 4u;
      async_ld16((unsigned)(size_t)&s_ah[buf][0] + ldso, A_hi, ga);
      async_ld16((unsigned)(size_t)&s_al[buf][0] + ldso, A_lo, ga);
      async_ld16((unsigned)(size_t)&s_bh[buf][0] + ldso, B_hi, gb);
      async_ld16((unsigned)(size_t)&s_bl[buf][0] + ldso, B_lo, gb);
    }
  };

  issue_stage(0, 0);

  for (int kt = 0; kt < kt_count; ++kt) {
    int cur = kt & 1;
    wait_async0();        // my 8 DMAs for tile kt have landed in LDS
    __syncthreads();      // everyone's tile kt landed; buffer cur^1 fully read

    if (kt + 1 < kt_count) issue_stage(kt + 1, cur ^ 1);  // overlap with WMMA

    if (kt + 2 < kt_count) {  // warm L2 for the tile after next
      unsigned pk = (unsigned)((kt + 2) * BK + seg * 8) * 2u;
      __builtin_prefetch((const char*)A_hi + (size_t)(m_blk + row0) * K * 2 + pk, 0, 1);
      __builtin_prefetch((const char*)B_hi + (size_t)(n_blk + row0) * K * 2 + pk, 0, 1);
    }

    const unsigned int* pah = &s_ah[cur][0];
    const unsigned int* pal = &s_al[cur][0];
    const unsigned int* pbh = &s_bh[cur][0];
    const unsigned int* pbl = &s_bl[cur][0];

    v16bf bh[2], bl[2];
#pragma unroll
    for (int nt = 0; nt < 2; ++nt) {
      int r = wn * 32 + nt * 16 + l16;
      bh[nt] = frag_b(pbh, r, half);
      bl[nt] = frag_b(pbl, r, half);
    }
#pragma unroll
    for (int mt = 0; mt < 4; ++mt) {
      int r = wm * 64 + mt * 16 + l16;
      v16bf ah = frag_a(pah, r, half);
      v16bf al = frag_a(pal, r, half);
#pragma unroll
      for (int nt = 0; nt < 2; ++nt) {
        acc[mt][nt] = __builtin_amdgcn_wmma_f32_16x16x32_bf16(
            false, ah, false, bh[nt], (short)0, acc[mt][nt], false, false);
        acc[mt][nt] = __builtin_amdgcn_wmma_f32_16x16x32_bf16(
            false, ah, false, bl[nt], (short)0, acc[mt][nt], false, false);
        acc[mt][nt] = __builtin_amdgcn_wmma_f32_16x16x32_bf16(
            false, al, false, bh[nt], (short)0, acc[mt][nt], false, false);
      }
    }
  }

  // Epilogue: C/D layout (ISA 7.12.2): VGPR r -> M = r + 8*half, N = l16.
#pragma unroll
  for (int nt = 0; nt < 2; ++nt) {
    int col = n_blk + wn * 32 + nt * 16 + l16;
    float bv = bias[col];
#pragma unroll
    for (int mt = 0; mt < 4; ++mt) {
      int mrow = m_blk + wm * 64 + mt * 16 + half * 8;
#pragma unroll
      for (int r = 0; r < 8; ++r)
        out[(size_t)(mrow + r) * N + col] = acc[mt][nt][r] + bv;
    }
  }
}

// ---------------------------------------------------------------------------
// Host launcher
// inputs: 0=batch_emb[N*D] f32, 1=seg[N] (unused: uniform NPER), 2=targets[B*2] int,
//         3=W[T*3D] f32, 4=b[T] f32. Output: [B*T] f32.
// ---------------------------------------------------------------------------
extern "C" void kernel_launch(void* const* d_in, const int* in_sizes, int n_in,
                              void* d_out, int out_size, void* d_ws, size_t ws_size,
                              hipStream_t stream) {
  const float* emb     = (const float*)d_in[0];
  const int*   targets = (const int*)d_in[2];
  const float* W       = (const float*)d_in[3];
  const float* bias    = (const float*)d_in[4];
  float*       out     = (float*)d_out;

  int T      = in_sizes[4];          // 2048
  int B      = in_sizes[2] / 2;      // 4096
  int K3     = in_sizes[3] / T;      // 3*D = 1536
  int D      = K3 / 3;               // 512
  int Nnodes = in_sizes[0] / D;      // 262144
  int NPER   = Nnodes / B;           // 64

  // Workspace partition (all 256B aligned): A_hi, A_lo [B x 3D] bf16;
  // W_hi, W_lo [T x 3D] bf16. Total ~38 MB.
  size_t offs = 0;
  auto take = [&](size_t bytes) {
    void* p = (char*)d_ws + offs;
    offs += (bytes + 255) & ~(size_t)255;
    return p;
  };
  __bf16* Ahi = (__bf16*)take((size_t)B * K3 * sizeof(__bf16));
  __bf16* Alo = (__bf16*)take((size_t)B * K3 * sizeof(__bf16));
  __bf16* Whi = (__bf16*)take((size_t)T * K3 * sizeof(__bf16));
  __bf16* Wlo = (__bf16*)take((size_t)T * K3 * sizeof(__bf16));

  build_final_kernel<<<B, 256, 0, stream>>>(emb, targets, Ahi, Alo, NPER, D);

  int nW = T * K3;
  int dgrid = (nW + 1023) / 1024;
  decompose_mat_kernel<<<dgrid, 256, 0, stream>>>(W, Whi, Wlo, nW);

  dim3 grid(T / BN, B / BM);   // (16, 32)
  gemm_bf16x3_kernel<<<grid, 256, 0, stream>>>(Ahi, Alo, Whi, Wlo, bias, out,
                                               B, T, K3);
}
